// SpectralConv2d_15710990369490
// MI455X (gfx1250) — matrix operand order repaired and checked by reading, not compiled, not run on hardware
//
#include <hip/hip_runtime.h>
#include <hip/hip_bf16.h>

// ---------------------------------------------------------------------------
// SpectralConv2d (FNO layer) for MI455X / gfx1250, wave32, all-WMMA f16->f32.
// B=16, CI=CO=64, H=W=256, M1=M2=20.
//   A) partial W-DFT (WMMA)          -> X1t f16 [48 col][262144 r] (transposed)
//   B) H-DFT, 40 kx modes (WMMA)     -> X2  f16 [800 mode][re,im,-im][1024 bi]
//   C) per-mode complex channel GEMM -> YS  f16 swizzled B-frags per (b,o)
//   D) inverse H-DFT (WMMA)          -> Y1  f16 [262144 r][64] (re|im|pad)
//   E) inverse W-DFT (WMMA)          -> out f32
// f16 WMMA has no A/B negation (ISA 7.12 NEG table), so negated twiddle /
// data planes are precomputed. Scales: weights *4096 (f16 range), Y *1/65536
// at stage C store, *1/4096 in f32 at stage D store.
// ---------------------------------------------------------------------------

typedef __attribute__((ext_vector_type(16))) _Float16 v16h;
typedef __attribute__((ext_vector_type(8)))  float    v8f;

#define NB   16
#define NCI  64
#define NCO  64
#define NH   256
#define NW   256
#define NM   20              // M1 == M2 == 20
#define NKX  40              // 20 positive + 20 negative kx modes
#define NMODE (NKX*NM)       // 800
#define NR   (NB*NCI*NH)     // 262144 rows for W-direction GEMMs

// ---- workspace layout (byte offsets, all 256-aligned) ----
#define OFF_EWS  ((size_t)0)          // f16 swz [8ks][3nt][32][16] fwd W twiddles
#define OFF_GS   ((size_t)24576)      // f16 swz [2ks][16nt][32][16] inv W twiddles
#define OFF_EHC  ((size_t)57344)      // f16 swz [8ks][3nt][32][16] fwd H cos
#define OFF_EHS  ((size_t)81920)      // f16 swz  ...               fwd H sin
#define OFF_EHN  ((size_t)106496)     // f16 swz  ...               fwd H -sin
#define OFF_EIC  ((size_t)131072)     // f16 plain [256 h][64 kx]   inv H cos
#define OFF_EIS  ((size_t)163840)     // f16 plain                  inv H sin
#define OFF_EIN  ((size_t)196608)     // f16 plain                  inv H -sin
#define OFF_WC   ((size_t)229376)     // f16 swz [800][2][2ks][4nt][32][16]
#define OFF_X1T  ((size_t)13336576)   // f16 [48 c][262144 r]
#define OFF_X2   ((size_t)38502400)   // f16 [800][3][1024]
#define OFF_YS   ((size_t)43417600)   // f16 [1024 bo][2 pl][2ks][2nt][32][16]
#define OFF_Y1   ((size_t)51806208)   // f16 [262144 r][64] (re j | im j | 0)

// ---------------------------------------------------------------------------
// WMMA fragment helpers (contiguous A/B k order, k = 16*(lane/16)+s, wave32)
// ---------------------------------------------------------------------------
__device__ __forceinline__ v16h load_a_frag(const _Float16* src, int ld) {
  // A 16x32 f16: lanes 0-15 M=lane, K={0..15}; lanes 16-31 M=lane-16,
  // K={16..31}. Two contiguous 16-byte runs per lane -> b128 loads.
  int lane = threadIdx.x & 31;
  int m  = lane & 15;
  int kb = (lane >> 4) << 4;
  const _Float16* r = src + (size_t)m * ld;
  v16h a;
#pragma unroll
  for (int p = 0; p < 8; ++p) {
    a[p]     = r[kb + p];
    a[8 + p] = r[8 + kb + p];
  }
  return a;
}

__device__ __forceinline__ v16h load_b_swz(const _Float16* base, int tile) {
  // Pre-swizzled B: lane's 16 halves contiguous, 32B aligned -> one v16h load.
  int lane = threadIdx.x & 31;
  return *(const v16h*)(base + ((size_t)tile * 32 + lane) * 16);
}

__device__ __forceinline__ void store_c_frag(float* dst, int ld, v8f c) {
  int lane = threadIdx.x & 31;
  int n  = lane & 15;
  int mb = (lane >> 4) << 3;
#pragma unroll
  for (int r = 0; r < 8; ++r) dst[(size_t)(mb + r) * ld + n] = c[r];
}

__device__ __forceinline__ void store_f16x8(_Float16* dst, v8f v, float scale) {
  // pack 8 f32 -> 8 f16 and store as one 16-byte vector (dst 16B aligned)
  union { _Float16 h[8]; uint4 u; } p;
#pragma unroll
  for (int r = 0; r < 8; ++r) p.h[r] = (_Float16)(v[r] * scale);
  *(uint4*)dst = p.u;
}

__device__ __forceinline__ v8f wmma16(v16h a, v16h b, v8f c) {
  return __builtin_amdgcn_wmma_f32_16x16x32_f16(false, a, false, b,
                                                (short)0, c, false, false);
}

// ---------------------------------------------------------------------------
// Init kernels (run every launch; deterministic)
// ---------------------------------------------------------------------------
__global__ void init_twiddles(_Float16* __restrict__ EwS, _Float16* __restrict__ GS,
                              _Float16* __restrict__ EhC, _Float16* __restrict__ EhS,
                              _Float16* __restrict__ EhN, _Float16* __restrict__ EiC,
                              _Float16* __restrict__ EiS, _Float16* __restrict__ EiN) {
  const float TWO_PI = 6.28318530717958647692f;
  int t = blockIdx.x * 256 + threadIdx.x;

  // EwS swizzled: B[k=w][c]; k = ks*32+(lane>>4)*16+i, c = nt*16+(lane&15)
  if (t < 12288) {
    int i = t & 15, lane = (t >> 4) & 31;
    int nt = (t >> 9) % 3, ks = t / 1536;
    int k = ks * 32 + ((lane >> 4) << 4) + i;
    int c = nt * 16 + (lane & 15);
    int j = c >> 1;
    float v = 0.f;
    if (c < 2 * NM) {
      float th = TWO_PI * (float)((k * j) & 255) * (1.0f / 256.0f);
      v = (c & 1) ? -__sinf(th) : __cosf(th);    // exp(-i th)
    }
    EwS[t] = (_Float16)v;
  }
  // GS swizzled: rows k<20 -> wt_j*cos (re part), 20<=k<40 -> -wt_j*sin (im)
  if (t < 16384) {
    int i = t & 15, lane = (t >> 4) & 31;
    int nt = (t >> 9) & 15, ks = t >> 13;
    int k = ks * 32 + ((lane >> 4) << 4) + i;
    int w = nt * 16 + (lane & 15);
    float v = 0.f;
    if (k < 2 * NM) {
      int j = (k < NM) ? k : (k - NM);
      float wt = (j == 0) ? 1.f : 2.f;
      float th = TWO_PI * (float)((w * j) & 255) * (1.0f / 256.0f);
      v = (k < NM) ? wt * __cosf(th) : -wt * __sinf(th);
    }
    GS[t] = (_Float16)v;
  }
  // Forward-H B planes, swizzled: B[k=h][c=kx]; zero for kx>=40
  if (t < 12288) {
    int i = t & 15, lane = (t >> 4) & 31;
    int nt = (t >> 9) % 3, ks = t / 1536;
    int h  = ks * 32 + ((lane >> 4) << 4) + i;
    int kx = nt * 16 + (lane & 15);
    float cv = 0.f, sv = 0.f;
    if (kx < NKX) {
      int kxf = (kx < NM) ? kx : (kx + 216);
      float th = TWO_PI * (float)((kxf * h) & 255) * (1.0f / 256.0f);
      cv = __cosf(th); sv = __sinf(th);
    }
    EhC[t] = (_Float16)cv;
    EhS[t] = (_Float16)sv;
    EhN[t] = (_Float16)(-sv);
  }
  // Inverse-H A planes, plain [h][64]; cols >=40 zero (K padding)
  if (t < 16384) {
    int h = t >> 6, c = t & 63;
    float cv = 0.f, sv = 0.f;
    if (c < NKX) {
      int kxf = (c < NM) ? c : (c + 216);
      float th = TWO_PI * (float)((kxf * h) & 255) * (1.0f / 256.0f);
      cv = __cosf(th); sv = __sinf(th);
    }
    EiC[t] = (_Float16)cv;
    EiS[t] = (_Float16)sv;
    EiN[t] = (_Float16)(-sv);
  }
}

__global__ void init_weights(const float* __restrict__ w1re, const float* __restrict__ w1im,
                             const float* __restrict__ w2re, const float* __restrict__ w2im,
                             _Float16* __restrict__ Wc) {
  int t = blockIdx.x * 256 + threadIdx.x;   // over 800*4096
  if (t >= NMODE * NCI * NCO) return;
  int mode = t >> 12;
  int io   = t & 4095;
  int i = io >> 6, o = io & 63;
  int kx = mode / NM, ky = mode % NM;
  int kxl = (kx < NM) ? kx : (kx - NM);
  size_t widx = (((size_t)i * NCO + o) * NM + kxl) * NM + ky;
  float re = (kx < NM) ? w1re[widx] : w2re[widx];
  float im = (kx < NM) ? w1im[widx] : w2im[widx];
  int ks = i >> 5, kk = i & 31;
  int lane = ((kk >> 4) << 4) | (o & 15);
  int nt = o >> 4, elem = kk & 15;
  size_t off = (((size_t)(ks * 4 + nt)) * 32 + lane) * 16 + elem;
  size_t base = (size_t)mode * 8192;
  Wc[base + off]        = (_Float16)(re * 4096.0f);
  Wc[base + 4096 + off] = (_Float16)(im * 4096.0f);
}

__global__ void zero_pads(uint4* __restrict__ ys, uint4* __restrict__ y1) {
  size_t t = (size_t)blockIdx.x * 256 + threadIdx.x;
  const uint4 z = {0u, 0u, 0u, 0u};
  if (t < 524288)  ys[t] = z;     // 8 MB YS
  if (t < 2097152) y1[t] = z;     // 32 MB Y1
}

// ---------------------------------------------------------------------------
// Stage A: X1t[c][r] = partial DFT along W of x (transposed f16 store)
//   M = 262144 (64 rows/block), N = 48 (3 tiles), K = 256 (8 steps)
// ---------------------------------------------------------------------------
__global__ void __launch_bounds__(384)
stageA(const float* __restrict__ x, const _Float16* __restrict__ EwS,
       _Float16* __restrict__ X1t) {
  __shared__ _Float16 sA[64][256];   // 32 KB, x tile in f16
  const int r0 = blockIdx.x * 64;

  for (int idx = threadIdx.x; idx < 64 * 256; idx += 384) {
    int row = idx >> 8, col = idx & 255;
    sA[row][col] = (_Float16)x[(size_t)(r0 + row) * NW + col];
  }
  __syncthreads();

  int wave = threadIdx.x >> 5;       // 12 waves
  int mt = wave / 3, nt = wave % 3;
  v8f acc = {};
#pragma unroll
  for (int ks = 0; ks < 8; ++ks) {
    v16h a = load_a_frag(&sA[mt * 16][ks * 32], 256);
    v16h b = load_b_swz(EwS, ks * 3 + nt);
    acc = wmma16(a, b, acc);
  }
  int lane = threadIdx.x & 31;
  int c  = nt * 16 + (lane & 15);
  int mb = (lane >> 4) << 3;
  // transposed store: 8 consecutive rows -> one packed 16B store
  store_f16x8(X1t + (size_t)c * NR + r0 + mt * 16 + mb, acc, 1.0f);
}

// ---------------------------------------------------------------------------
// Stage B: H-DFT as WMMA GEMM per (ky, bi-tile).
//   A = X1t col planes [64 bi x 256 h], B = swizzled cos/sin/-sin [256 x 48]
//   re = Xre*cos + Xim*sin ; im = Xim*cos + Xre*(-sin)
// ---------------------------------------------------------------------------
__global__ void __launch_bounds__(384)
stageB(const _Float16* __restrict__ X1t, const _Float16* __restrict__ EhC,
       const _Float16* __restrict__ EhS, const _Float16* __restrict__ EhN,
       _Float16* __restrict__ X2) {
  int ky = blockIdx.x / 16;
  int bt = blockIdx.x % 16;
  int wave = threadIdx.x >> 5;       // 12 waves
  int mt = wave / 3, nt = wave % 3;
  int biBase = bt * 64 + mt * 16;
  const _Float16* Are = X1t + (size_t)(2 * ky) * NR + (size_t)biBase * 256;
  const _Float16* Aim = Are + NR;

  v8f accR = {}, accI = {};
#pragma unroll
  for (int ks = 0; ks < 8; ++ks) {
    v16h are = load_a_frag(Are + ks * 32, 256);
    v16h aim = load_a_frag(Aim + ks * 32, 256);
    v16h bc = load_b_swz(EhC, ks * 3 + nt);
    v16h bs = load_b_swz(EhS, ks * 3 + nt);
    v16h bn = load_b_swz(EhN, ks * 3 + nt);
    accR = wmma16(are, bc, accR);
    accR = wmma16(aim, bs, accR);
    accI = wmma16(aim, bc, accI);
    accI = wmma16(are, bn, accI);
  }
  int lane = threadIdx.x & 31;
  int n = lane & 15, mb = (lane >> 4) << 3;
  int kx = nt * 16 + n;
  if (kx < NKX) {
    size_t base = ((size_t)(kx * NM + ky)) * 3072 + biBase + mb;
    store_f16x8(X2 + base,        accR,  1.0f);
    store_f16x8(X2 + base + 1024, accI,  1.0f);
    store_f16x8(X2 + base + 2048, accI, -1.0f);   // f16 WMMA has no A/B neg
  }
}

// ---------------------------------------------------------------------------
// Stage C: per-mode complex channel GEMM; output scattered into swizzled
// B-fragments per (b,o) for stage D (scale 1/65536 folded in).
// ---------------------------------------------------------------------------
__global__ void __launch_bounds__(128)
stageC(const _Float16* __restrict__ X2, const _Float16* __restrict__ Wc,
       _Float16* __restrict__ YS) {
  int mode = blockIdx.x;
  int wave = threadIdx.x >> 5;       // 0..3 -> N tile (o)
  const _Float16* Xre  = X2 + (size_t)mode * 3072;
  const _Float16* Xim  = Xre + 1024;
  const _Float16* Xni  = Xre + 2048;
  const _Float16* Wre  = Wc + (size_t)mode * 8192;
  const _Float16* Wim  = Wre + 4096;

  v8f ar = {}, ai = {};
#pragma unroll
  for (int ks = 0; ks < 2; ++ks) {
    v16h xre = load_a_frag(Xre + ks * 32, 64);
    v16h xim = load_a_frag(Xim + ks * 32, 64);
    v16h xni = load_a_frag(Xni + ks * 32, 64);
    v16h wre = load_b_swz(Wre, ks * 4 + wave);
    v16h wim = load_b_swz(Wim, ks * 4 + wave);
    ar = wmma16(xre, wre, ar);
    ar = wmma16(xni, wim, ar);
    ai = wmma16(xre, wim, ai);
    ai = wmma16(xim, wre, ai);
  }
  // scatter into per-(b,o) swizzled B fragment for stage D
  int kx = mode / NM, ky = mode % NM;
  int ksd  = kx >> 5, elem = kx & 15;
  int laneT = (((kx & 31) >> 4) << 4) | (ky & 15);
  int ntd  = ky >> 4;
  size_t off = (((size_t)(ksd * 2 + ntd)) * 32 + laneT) * 16 + elem;
  int lane = threadIdx.x & 31;
  int o = wave * 16 + (lane & 15);
  int mb = (lane >> 4) << 3;
  const float sc = 1.0f / 65536.0f;
#pragma unroll
  for (int r = 0; r < 8; ++r) {
    size_t bo = (size_t)(mb + r) * 64 + o;
    YS[bo * 4096 + off]        = (_Float16)(ar[r] * sc);
    YS[bo * 4096 + 2048 + off] = (_Float16)(ai[r] * sc);
  }
}

// ---------------------------------------------------------------------------
// Stage D: inverse H-DFT as WMMA GEMM per (b,o).
//   A = cos/sin/-sin [256 h x 64 kx-pad], B = YS fragments [64 x 32]
//   re = cos*yr + (-sin)*yi ; im = cos*yi + sin*yr ; *1/4096 in f32 at store
// ---------------------------------------------------------------------------
__global__ void __launch_bounds__(256)
stageD(const _Float16* __restrict__ YS, const _Float16* __restrict__ EiC,
       const _Float16* __restrict__ EiS, const _Float16* __restrict__ EiN,
       _Float16* __restrict__ Y1) {
  int bo = blockIdx.x >> 1;
  int mh = blockIdx.x & 1;
  int wave = threadIdx.x >> 5;       // 8 waves
  int h0 = mh * 128 + wave * 16;
  const _Float16* Bre = YS + (size_t)bo * 4096;
  const _Float16* Bim = Bre + 2048;

  v16h ac0 = load_a_frag(EiC + h0 * 64, 64);
  v16h ac1 = load_a_frag(EiC + h0 * 64 + 32, 64);
  v16h as0 = load_a_frag(EiS + h0 * 64, 64);
  v16h as1 = load_a_frag(EiS + h0 * 64 + 32, 64);
  v16h an0 = load_a_frag(EiN + h0 * 64, 64);
  v16h an1 = load_a_frag(EiN + h0 * 64 + 32, 64);

  int lane = threadIdx.x & 31;
  int n = lane & 15, mb = (lane >> 4) << 3;
#pragma unroll
  for (int nt = 0; nt < 2; ++nt) {
    v16h br0 = load_b_swz(Bre, nt);
    v16h bi0 = load_b_swz(Bim, nt);
    v16h br1 = load_b_swz(Bre, 2 + nt);
    v16h bi1 = load_b_swz(Bim, 2 + nt);
    v8f aR = {}, aI = {};
    aR = wmma16(ac0, br0, aR); aR = wmma16(an0, bi0, aR);
    aR = wmma16(ac1, br1, aR); aR = wmma16(an1, bi1, aR);
    aI = wmma16(ac0, bi0, aI); aI = wmma16(as0, br0, aI);
    aI = wmma16(ac1, bi1, aI); aI = wmma16(as1, br1, aI);
    int j = nt * 16 + n;
    if (j < NM) {
      const float inv = 1.0f / 4096.0f;
      _Float16* base = Y1 + ((size_t)bo * NH + h0 + mb) * 64;
#pragma unroll
      for (int r = 0; r < 8; ++r) {
        base[(size_t)r * 64 + j]      = (_Float16)(aR[r] * inv);
        base[(size_t)r * 64 + NM + j] = (_Float16)(aI[r] * inv);
      }
    }
  }
}

// ---------------------------------------------------------------------------
// Stage E: out[r, 0..255] = Y1[r, 0..63] * G[64][256].  WMMA GEMM
// ---------------------------------------------------------------------------
__global__ void __launch_bounds__(256)
stageE(const _Float16* __restrict__ Y1, const _Float16* __restrict__ GS,
       float* __restrict__ out) {
  int mt = blockIdx.x;               // 0..16383
  int wave = threadIdx.x >> 5;       // 0..7
  const _Float16* A0 = Y1 + (size_t)mt * 16 * 64;
  v16h a0 = load_a_frag(A0, 64);
  v16h a1 = load_a_frag(A0 + 32, 64);
#pragma unroll
  for (int half = 0; half < 2; ++half) {
    int nt = wave * 2 + half;        // 0..15
    v8f acc = {};
    acc = wmma16(a0, load_b_swz(GS, nt), acc);
    acc = wmma16(a1, load_b_swz(GS, 16 + nt), acc);
    store_c_frag(out + (size_t)mt * 16 * 256 + nt * 16, 256, acc);
  }
}

// ---------------------------------------------------------------------------
extern "C" void kernel_launch(void* const* d_in, const int* in_sizes, int n_in,
                              void* d_out, int out_size, void* d_ws, size_t ws_size,
                              hipStream_t stream) {
  const float* x    = (const float*)d_in[0];
  const float* w1re = (const float*)d_in[1];
  const float* w1im = (const float*)d_in[2];
  const float* w2re = (const float*)d_in[3];
  const float* w2im = (const float*)d_in[4];
  float* out = (float*)d_out;

  char* ws = (char*)d_ws;
  _Float16* EwS = (_Float16*)(ws + OFF_EWS);
  _Float16* GS  = (_Float16*)(ws + OFF_GS);
  _Float16* EhC = (_Float16*)(ws + OFF_EHC);
  _Float16* EhS = (_Float16*)(ws + OFF_EHS);
  _Float16* EhN = (_Float16*)(ws + OFF_EHN);
  _Float16* EiC = (_Float16*)(ws + OFF_EIC);
  _Float16* EiS = (_Float16*)(ws + OFF_EIS);
  _Float16* EiN = (_Float16*)(ws + OFF_EIN);
  _Float16* Wc  = (_Float16*)(ws + OFF_WC);
  _Float16* X1t = (_Float16*)(ws + OFF_X1T);
  _Float16* X2  = (_Float16*)(ws + OFF_X2);
  _Float16* YS  = (_Float16*)(ws + OFF_YS);
  _Float16* Y1  = (_Float16*)(ws + OFF_Y1);

  init_twiddles<<<64, 256, 0, stream>>>(EwS, GS, EhC, EhS, EhN, EiC, EiS, EiN);
  init_weights<<<(NMODE * NCI * NCO + 255) / 256, 256, 0, stream>>>(
      w1re, w1im, w2re, w2im, Wc);
  zero_pads<<<8192, 256, 0, stream>>>((uint4*)YS, (uint4*)Y1);

  stageA<<<NR / 64, 384, 0, stream>>>(x, EwS, X1t);
  stageB<<<NM * 16, 384, 0, stream>>>(X1t, EhC, EhS, EhN, X2);
  stageC<<<NMODE, 128, 0, stream>>>(X2, Wc, YS);
  stageD<<<NB * NCO * 2, 256, 0, stream>>>(YS, EiC, EiS, EiN, Y1);
  stageE<<<NR / 16, 256, 0, stream>>>(Y1, GS, out);
}
